// RPNHead_2559800508425
// MI455X (gfx1250) — matrix operand order repaired and checked by reading, not compiled, not run on hardware
//
#include <hip/hip_runtime.h>
#include <hip/hip_bf16.h>

typedef __attribute__((ext_vector_type(16))) __bf16 v16bf;
typedef __attribute__((ext_vector_type(8)))  float  v8f;

union BfFrag { unsigned int u[8]; v16bf v; };

__device__ __forceinline__ unsigned short f2bf(float f) {
    unsigned int u = __float_as_uint(f);
    u += 0x7FFFu + ((u >> 16) & 1u);       // round-to-nearest-even
    return (unsigned short)(u >> 16);
}

// Two async 16B global->LDS copies (32B per lane), tracked on ASYNCcnt (+2 per wave).
__device__ __forceinline__ void async_copy_32B(unsigned lds_addr, const void* gaddr) {
    asm volatile(
        "global_load_async_to_lds_b128 %0, %1, off\n\t"
        "global_load_async_to_lds_b128 %0, %1, off offset:16"
        :: "v"(lds_addr), "v"(gaddr) : "memory");
}
// Wait until this wave's outstanding async ops <= 4 (i.e. only the newest batch).
__device__ __forceinline__ void wait_async_le4() {
    asm volatile("s_wait_asynccnt 0x4" ::: "memory");
}
__device__ __forceinline__ void wait_async_le0() {
    asm volatile("s_wait_asynccnt 0x0" ::: "memory");
}
__device__ __forceinline__ unsigned lds_addr_of(const void* p) {
    return (unsigned)(uintptr_t)p;   // flat shared addr: low 32 bits == LDS byte offset
}

// ---------------------------------------------------------------------------
// Prologue kernels: convert operands to bf16 GEMM layouts
// ---------------------------------------------------------------------------
__global__ void prep_zero(unsigned int* __restrict__ zp) {
    zp[threadIdx.x] = 0u;                            // 32 threads -> 128 B zero page
}

__global__ void prep_x(const float* __restrict__ x, unsigned short* __restrict__ xbf) {
    int i = blockIdx.x * 256 + threadIdx.x;          // one float4 per thread
    const float4 p = ((const float4*)x)[i];          // 33554432/4 threads, exact
    unsigned lo = (unsigned)f2bf(p.x) | ((unsigned)f2bf(p.y) << 16);
    unsigned hi = (unsigned)f2bf(p.z) | ((unsigned)f2bf(p.w) << 16);
    ((uint2*)xbf)[i] = make_uint2(lo, hi);
}

__global__ void prep_wshared(const float* __restrict__ w,      // [3,3,256,512] HWIO
                             unsigned short* __restrict__ Wt)  // [512][2304]  (n-major)
{
    int i = blockIdx.x * 256 + threadIdx.x;
    if (i >= 512 * 2304) return;
    int n = i / 2304, k = i % 2304;
    Wt[i] = f2bf(w[k * 512 + n]);
}

__global__ void prep_wheads(const float* __restrict__ wc,      // [512][30]
                            const float* __restrict__ wr,      // [512][60]
                            unsigned short* __restrict__ Wt2)  // [96][512]
{
    int i = blockIdx.x * 256 + threadIdx.x;
    if (i >= 96 * 512) return;
    int n = i / 512, k = i % 512;
    float v = (n < 30) ? wc[k * 30 + n] : ((n < 90) ? wr[k * 60 + (n - 30)] : 0.0f);
    Wt2[i] = f2bf(v);
}

// ---------------------------------------------------------------------------
// Kernel A: shared 3x3 conv as im2col GEMM  (M=131072, K=2304, N=512), bf16 WMMA
// Tile 128x128, 8 waves (2M x 4N), per wave 4x2 accumulators of 16x16.
// Double-buffered LDS; tiles streamed with GLOBAL_LOAD_ASYNC_TO_LDS_B128 that
// overlap the WMMAs (uniform 4 async issues/wave/iter; OOB rows read a zero page).
// Epilogue: +bias, relu6, store bf16 "shared" activation.
// ---------------------------------------------------------------------------
__global__ __launch_bounds__(256) void conv3x3_gemm(
    const unsigned short* __restrict__ xbf,   // [8,128,128,256] bf16
    const unsigned short* __restrict__ Wt,    // [512][2304] bf16
    const float* __restrict__ bsh,            // [512]
    const unsigned short* __restrict__ zpage, // 128 B of zeros
    unsigned short* __restrict__ shout)       // [131072][512] bf16
{
    __shared__ __align__(16) unsigned short As[2][128 * 32];
    __shared__ __align__(16) unsigned short Bs[2][128 * 32];

    const int t    = threadIdx.x;
    const int lane = t & 31;
    const int wave = t >> 5;
    const int wM   = wave >> 2;          // 0..1  -> 64 M rows each
    const int wN   = wave & 3;           // 0..3  -> 32 N cols each

    const int m0 = blockIdx.x * 128;
    const int n0 = blockIdx.y * 128;

    const int lr = t >> 1;               // 0..127 tile row loaded by this thread
    const int lh = t & 1;                // half (16 elements)

    // pixel coords for A-tile row lr
    const int m  = m0 + lr;
    const int bb = m >> 14;
    const int yy = (m >> 7) & 127;
    const int xx = m & 127;

    v8f acc[4][2];
    #pragma unroll
    for (int i = 0; i < 4; i++)
        #pragma unroll
        for (int j = 0; j < 2; j++)
            #pragma unroll
            for (int e = 0; e < 8; e++) acc[i][j][e] = 0.0f;

    const int khalfA = (lane >> 4) << 3;     // 0 / 8
    const int rowA0  = (wM << 6) + (lane & 15);
    const int khalfB = (lane >> 4) << 3;     // 0 / 8
    const int colB0  = (wN << 5) + (lane & 15);

    const unsigned laA[2] = { lds_addr_of(As[0] + lr * 32 + lh * 16),
                              lds_addr_of(As[1] + lr * 32 + lh * 16) };
    const unsigned laB[2] = { lds_addr_of(Bs[0] + lr * 32 + lh * 16),
                              lds_addr_of(Bs[1] + lr * 32 + lh * 16) };

    // im2col source address for K chunk kq (branchless: OOB -> zero page)
    auto a_src = [&](int kq) -> const void* {
        const int f   = kq >> 8;             // 0..8 filter tap
        const int fy  = f / 3 - 1;
        const int fx  = f - (f / 3) * 3 - 1;
        const int iy  = yy + fy;
        const int ix  = xx + fx;
        const bool inb = ((unsigned)iy < 128u) & ((unsigned)ix < 128u);
        const int iyc = min(max(iy, 0), 127);
        const int ixc = min(max(ix, 0), 127);
        const size_t off = ((size_t)((bb << 7) + iyc) * 128 + ixc) * 256
                         + ((kq & 255) + lh * 16);
        return inb ? (const void*)(xbf + off) : (const void*)zpage;
    };
    auto b_src = [&](int kq) -> const void* {
        return (const void*)(Wt + (size_t)(n0 + lr) * 2304 + kq + lh * 16);
    };

    // prefetch K chunk 0 into buffer 0
    async_copy_32B(laA[0], a_src(0));
    async_copy_32B(laB[0], b_src(0));

    int buf = 0;
    for (int kk = 0; kk < 2304; kk += 32, buf ^= 1) {
        if (kk + 32 < 2304) {
            async_copy_32B(laA[buf ^ 1], a_src(kk + 32));   // stream next tile
            async_copy_32B(laB[buf ^ 1], b_src(kk + 32));
            wait_async_le4();                               // current tile landed
        } else {
            wait_async_le0();
        }
        __syncthreads();

        // ---- fragments + WMMA (current buffer) ----
        BfFrag bfr[2];
        #pragma unroll
        for (int nj = 0; nj < 2; nj++) {
            const unsigned short* bp = Bs[buf] + (colB0 + nj * 16) * 32 + khalfB;
            #pragma unroll
            for (int v = 0; v < 8; v++) bfr[nj].u[v] = *(const unsigned int*)(bp + 2 * (v & 3) + ((v >= 4) ? 16 : 0));
        }
        #pragma unroll
        for (int mi = 0; mi < 4; mi++) {
            BfFrag a;
            const unsigned short* ap = As[buf] + (rowA0 + mi * 16) * 32;
            #pragma unroll
            for (int v = 0; v < 8; v++) {
                const int koff = 2 * (v & 3) + khalfA + ((v >= 4) ? 16 : 0);
                a.u[v] = *(const unsigned int*)(ap + koff);
            }
            #pragma unroll
            for (int nj = 0; nj < 2; nj++)
                acc[mi][nj] = __builtin_amdgcn_wmma_f32_16x16x32_bf16(
                    false, a.v, false, bfr[nj].v, (short)0, acc[mi][nj], false, false);
        }
        __syncthreads();   // all waves done reading buf before it is overwritten
    }

    // ---- epilogue: bias + relu6 -> bf16 ----
    #pragma unroll
    for (int nj = 0; nj < 2; nj++) {
        const int n = n0 + (wN << 5) + nj * 16 + (lane & 15);
        const float bias = bsh[n];
        #pragma unroll
        for (int mi = 0; mi < 4; mi++) {
            #pragma unroll
            for (int v = 0; v < 8; v++) {
                const int mm = m0 + (wM << 6) + mi * 16 + v + ((lane >> 4) << 3);
                float val = acc[mi][nj][v] + bias;
                val = fminf(fmaxf(val, 0.0f), 6.0f);
                shout[(size_t)mm * 512 + n] = f2bf(val);
            }
        }
    }
}

// ---------------------------------------------------------------------------
// Kernel B: head 1x1 convs as GEMM (M=131072, K=512, N=96[30 cls | 60 reg | pad])
// Tile 128x96, 8 waves (4M x 2N), per wave 2x3 accumulators. Double-buffered
// async staging; B tile padded to 128 rows (zero page) for uniform issue.
// Epilogue: +bias; cls -> logits + pairwise softmax; reg -> deltas.
// ---------------------------------------------------------------------------
__global__ __launch_bounds__(256) void heads_gemm(
    const unsigned short* __restrict__ shin,   // [131072][512] bf16
    const unsigned short* __restrict__ Wt2,    // [96][512] bf16
    const float* __restrict__ bcls,            // [30]
    const float* __restrict__ breg,            // [60]
    const unsigned short* __restrict__ zpage,  // 128 B of zeros
    float* __restrict__ out_logits,            // [131072][30]
    float* __restrict__ out_probs,             // [131072][30]
    float* __restrict__ out_deltas)            // [131072][60]
{
    __shared__ __align__(16) unsigned short As[2][128 * 32];
    __shared__ __align__(16) unsigned short Bs[2][128 * 32];   // rows 96..127 = pad

    const int t    = threadIdx.x;
    const int lane = t & 31;
    const int wave = t >> 5;
    const int wM   = wave >> 1;          // 0..3 -> 32 M rows each
    const int wN   = wave & 1;           // 0..1 -> 48 N cols each
    const int m0   = blockIdx.x * 128;

    const int lr = t >> 1;
    const int lh = t & 1;

    v8f acc[2][3];
    #pragma unroll
    for (int i = 0; i < 2; i++)
        #pragma unroll
        for (int j = 0; j < 3; j++)
            #pragma unroll
            for (int e = 0; e < 8; e++) acc[i][j][e] = 0.0f;

    const int khalfA = (lane >> 4) << 3;
    const int rowA0  = (wM << 5) + (lane & 15);
    const int khalfB = (lane >> 4) << 3;
    const int colB0  = wN * 48 + (lane & 15);

    const unsigned laA[2] = { lds_addr_of(As[0] + lr * 32 + lh * 16),
                              lds_addr_of(As[1] + lr * 32 + lh * 16) };
    const unsigned laB[2] = { lds_addr_of(Bs[0] + lr * 32 + lh * 16),
                              lds_addr_of(Bs[1] + lr * 32 + lh * 16) };

    auto a_src = [&](int kq) -> const void* {
        return (const void*)(shin + (size_t)(m0 + lr) * 512 + kq + lh * 16);
    };
    auto b_src = [&](int kq) -> const void* {
        const size_t off = (size_t)lr * 512 + kq + lh * 16;
        return (lr < 96) ? (const void*)(Wt2 + off) : (const void*)zpage;
    };

    async_copy_32B(laA[0], a_src(0));
    async_copy_32B(laB[0], b_src(0));

    int buf = 0;
    for (int kk = 0; kk < 512; kk += 32, buf ^= 1) {
        if (kk + 32 < 512) {
            async_copy_32B(laA[buf ^ 1], a_src(kk + 32));
            async_copy_32B(laB[buf ^ 1], b_src(kk + 32));
            wait_async_le4();
        } else {
            wait_async_le0();
        }
        __syncthreads();

        BfFrag bfr[3];
        #pragma unroll
        for (int nj = 0; nj < 3; nj++) {
            const unsigned short* bp = Bs[buf] + (colB0 + nj * 16) * 32 + khalfB;
            #pragma unroll
            for (int v = 0; v < 8; v++) bfr[nj].u[v] = *(const unsigned int*)(bp + 2 * (v & 3) + ((v >= 4) ? 16 : 0));
        }
        #pragma unroll
        for (int mi = 0; mi < 2; mi++) {
            BfFrag a;
            const unsigned short* ap = As[buf] + (rowA0 + mi * 16) * 32;
            #pragma unroll
            for (int v = 0; v < 8; v++) {
                const int koff = 2 * (v & 3) + khalfA + ((v >= 4) ? 16 : 0);
                a.u[v] = *(const unsigned int*)(ap + koff);
            }
            #pragma unroll
            for (int nj = 0; nj < 3; nj++)
                acc[mi][nj] = __builtin_amdgcn_wmma_f32_16x16x32_bf16(
                    false, a.v, false, bfr[nj].v, (short)0, acc[mi][nj], false, false);
        }
        __syncthreads();
    }

    // ---- epilogue: bias, softmax pairs (adjacent lanes hold n and n^1 for same m) ----
    #pragma unroll
    for (int nj = 0; nj < 3; nj++) {
        const int n = wN * 48 + nj * 16 + (lane & 15);
        const float bias = (n < 30) ? bcls[n] : ((n < 90) ? breg[n - 30] : 0.0f);
        #pragma unroll
        for (int mi = 0; mi < 2; mi++) {
            #pragma unroll
            for (int v = 0; v < 8; v++) {
                const int mm = m0 + (wM << 5) + mi * 16 + v + ((lane >> 4) << 3);
                const float val = acc[mi][nj][v] + bias;
                const float partner = __shfl_xor(val, 1, 32);  // other logit of the pair
                if (n < 30) {
                    out_logits[(size_t)mm * 30 + n] = val;
                    const float mx = fmaxf(val, partner);
                    const float e0 = __expf(val - mx);
                    const float e1 = __expf(partner - mx);
                    out_probs[(size_t)mm * 30 + n] = e0 / (e0 + e1);
                } else if (n < 90) {
                    out_deltas[(size_t)mm * 60 + (n - 30)] = val;
                }
            }
        }
    }
}

// ---------------------------------------------------------------------------
extern "C" void kernel_launch(void* const* d_in, const int* in_sizes, int n_in,
                              void* d_out, int out_size, void* d_ws, size_t ws_size,
                              hipStream_t stream) {
    const float* x        = (const float*)d_in[0];
    const float* w_shared = (const float*)d_in[1];
    const float* b_shared = (const float*)d_in[2];
    const float* w_cls    = (const float*)d_in[3];
    const float* b_cls    = (const float*)d_in[4];
    const float* w_reg    = (const float*)d_in[5];
    const float* b_reg    = (const float*)d_in[6];

    char* ws = (char*)d_ws;
    unsigned short* Wt  = (unsigned short*)ws;                      // 512*2304 bf16
    unsigned short* Wt2 = (unsigned short*)(ws + 2359296);          // 96*512 bf16
    unsigned short* zpg = (unsigned short*)(ws + 2457600);          // 128 B zero page
    unsigned short* xbf = (unsigned short*)(ws + 2457856);          // 8*128*128*256 bf16
    unsigned short* sh  = (unsigned short*)(ws + 69566720);         // 131072*512 bf16

    prep_zero   <<<1, 32, 0, stream>>>((unsigned int*)zpg);
    prep_x      <<<32768, 256, 0, stream>>>(x, xbf);                // 33.5M elems, 4/thread
    prep_wshared<<<(512 * 2304 + 255) / 256, 256, 0, stream>>>(w_shared, Wt);
    prep_wheads <<<(96 * 512 + 255) / 256, 256, 0, stream>>>(w_cls, w_reg, Wt2);

    dim3 gA(1024, 4);   // M/128 x N/128
    conv3x3_gemm<<<gA, 256, 0, stream>>>(xbf, Wt, b_shared, zpg, sh);

    float* out0 = (float*)d_out;            // logits [8*245760*2]
    float* out1 = out0 + 3932160;           // probs
    float* out2 = out0 + 7864320;           // deltas [8*245760*4]
    heads_gemm<<<1024, 256, 0, stream>>>(sh, Wt2, b_cls, b_reg, zpg, out0, out1, out2);
}
